// CrossAttention_7069516169298
// MI455X (gfx1250) — matrix operand order repaired and checked by reading, not compiled, not run on hardware
//
#include <hip/hip_runtime.h>

#define BB 2
#define CC 128
#define NN 4096      // H*W
#define NH 8
#define DH 16
#define GG 8
#define NSCALE 0.25f // dh^-0.5
#define GN_EPS 1e-5f

typedef __attribute__((ext_vector_type(16))) _Float16 f16x16;
typedef __attribute__((ext_vector_type(8)))  float    f32x8;

union FragAB { f16x16 v; unsigned int u[8]; };

#if __has_builtin(__builtin_amdgcn_global_load_async_to_lds_b128) && \
    __has_builtin(__builtin_amdgcn_s_wait_asynccnt)
#define HAVE_ASYNC_LDS 1
#define GLOBAL_AS __attribute__((address_space(1)))
#define LDS_AS    __attribute__((address_space(3)))
typedef int i32x4 __attribute__((vector_size(16)));   // matches builtin param type
typedef GLOBAL_AS i32x4* gvec4p;
typedef LDS_AS    i32x4* lvec4p;
#else
#define HAVE_ASYNC_LDS 0
#endif

static __device__ __forceinline__ unsigned short f2h_bits(float f) {
  _Float16 h = (_Float16)f;
  unsigned short u;
  __builtin_memcpy(&u, &h, 2);
  return u;
}

// ---------------- k0: convert Wq/Wk/Wv/Wo fp32 -> f16 ----------------
__global__ void k0_convert_weights(const float* __restrict__ Wq, const float* __restrict__ Wk,
                                   const float* __restrict__ Wv, const float* __restrict__ Wo,
                                   unsigned short* __restrict__ wb) {
  int i = blockIdx.x * blockDim.x + threadIdx.x;     // 0..65535
  const int M = CC * CC;                             // 16384
  const float* src = (i < M) ? Wq : (i < 2*M) ? Wk : (i < 3*M) ? Wv : Wo;
  wb[i] = f2h_bits(src[i & (M - 1)]);
}

// ---------------- k1: QKV projection (WMMA) ----------------
// Q,K stored [b,h,N,dh] f16 ; V stored transposed [b,h,dh,N] f16
__global__ void k1_qkv(const float* __restrict__ x,
                       const unsigned short* __restrict__ wWb,
                       const float* __restrict__ bq, const float* __restrict__ bk,
                       const float* __restrict__ bv,
                       unsigned short* __restrict__ wQ, unsigned short* __restrict__ wK,
                       unsigned short* __restrict__ wVt) {
  __shared__ __align__(16) unsigned short ldsx[16 * CC];   // [tok][c] f16
  const int bid = blockIdx.x;                // B * N/16
  const int b  = bid / (NN / 16);
  const int tb = (bid % (NN / 16)) * 16;
  const int t  = threadIdx.x;                // 0..127
  for (int i = 0; i < 16; ++i) {
    int idx = i * 128 + t;                   // 2048 elements
    int c = idx >> 4, tok = idx & 15;
    ldsx[tok * CC + c] = f2h_bits(x[((size_t)b * CC + c) * NN + tb + tok]);
  }
  __syncthreads();
  const unsigned int* ldsx32 = (const unsigned int*)ldsx;
  const unsigned int* wb32   = (const unsigned int*)wWb;
  const int wave = t >> 5, lane = t & 31;
  const int half = lane >> 4, l15 = lane & 15;

  for (int task = wave; task < 24; task += 4) {
    const int mat = task >> 3;               // 0=Q 1=K 2=V
    const int dt  = task & 7;                // head == d-tile
    const int dcol = dt * 16 + l15;          // B/D column (output dim)
    f32x8 acc = {};
    FragAB a, bf;
    for (int kk = 0; kk < 4; ++kk) {
      const int cA = kk * 32 + half * 16;
      #pragma unroll
      for (int v = 0; v < 4; ++v) {
        a.u[v]     = ldsx32[(l15 * CC + cA + 2 * v) >> 1];
        a.u[v + 4] = ldsx32[(l15 * CC + cA + 8 + 2 * v) >> 1];
      }
      const int cB = kk * 32 + half * 16;
      const unsigned int* wmat = wb32 + (size_t)mat * (CC * CC / 2) + dcol * (CC / 2);
      #pragma unroll
      for (int v = 0; v < 8; ++v) bf.u[v] = wmat[(cB >> 1) + v];
      acc = __builtin_amdgcn_wmma_f32_16x16x32_f16(false, a.v, false, bf.v,
                                                   (short)0, acc, false, false);
    }
    const float* bias = (mat == 0) ? bq : (mat == 1) ? bk : bv;
    const float bval = bias[dcol];
    const int bh = b * NH + dt;
    if (mat == 2) {                                  // wave-uniform branch
      unsigned short* dst = wVt + ((size_t)bh * DH + l15) * NN + tb + half * 8;
      #pragma unroll
      for (int j = 0; j < 8; ++j) dst[j] = f2h_bits(acc[j] + bval);
    } else {
      unsigned short* dst = (mat ? wK : wQ) + ((size_t)bh * NN + tb + half * 8) * DH + l15;
      #pragma unroll
      for (int j = 0; j < 8; ++j) dst[(size_t)j * DH] = f2h_bits(acc[j] + bval);
    }
  }
}

// ---------------- k2: flash attention, 64 keys/step, async double-buffered LDS ----------------
__global__ void k2_attn(const unsigned short* __restrict__ wQ,
                        const unsigned short* __restrict__ wK,
                        const unsigned short* __restrict__ wVt,
                        unsigned short* __restrict__ wAttn) {
  __shared__ __align__(16) unsigned int ldsK[2][512];   // [buf][key(64)][dh pair(8)]
  __shared__ __align__(16) unsigned int ldsV[2][512];   // [buf][dh(16)][key pair(32)]
  __shared__ __align__(16) unsigned int ldsP[4][512];   // per-wave probs [row16][kp32]
  __shared__ __align__(16) unsigned int ldsZ[8];        // zeros for K-frag upper lanes

  const int QG  = NN / 64;
  const int bid = blockIdx.x;
  const int bh  = bid / QG;
  const int qg  = bid % QG;
  const int t = threadIdx.x;
  const int wave = t >> 5, lane = t & 31;
  const int half = lane >> 4, l15 = lane & 15;
  const int qt = qg * 64 + wave * 16;
  const unsigned int* q32 = (const unsigned int*)wQ;

  if (t < 8) ldsZ[t] = 0u;

  // per-thread staging coordinates (16B of K, 16B of V per 64-key tile)
  const int vd = t >> 3, vchunk = t & 7;                 // V: row dh, 16B chunk
  const unsigned short* gKbase = wK  + (size_t)bh * NN * DH + t * 8;
  const unsigned short* gVbase = wVt + ((size_t)bh * DH + vd) * NN + vchunk * 8;

  FragAB qf;   // Q as A-operand: dh(16) padded to K=32 with zeros
  {
    const size_t qb = ((size_t)bh * NN + qt + l15) * 8;
    #pragma unroll
    for (int v = 0; v < 4; ++v) { qf.u[v] = half ? 0u : q32[qb + v]; qf.u[v + 4] = half ? 0u : q32[qb + 4 + v]; }
  }

  f32x8 acc = {};
  float mrow[8], lrow[8];
  #pragma unroll
  for (int j = 0; j < 8; ++j) { mrow[j] = -1e30f; lrow[j] = 0.0f; }

  unsigned short* ldsP16 = (unsigned short*)&ldsP[wave][0];
  const unsigned int* ldsPw = &ldsP[wave][0];

#if HAVE_ASYNC_LDS
#define STAGE_TILE(buf, kb_)                                                          \
  do {                                                                                \
    const unsigned short* gK = gKbase + (size_t)(kb_) * DH;                           \
    const unsigned short* gV = gVbase + (kb_);                                        \
    unsigned int* lK = &ldsK[buf][t * 4];                                             \
    unsigned int* lV = &ldsV[buf][vd * 32 + vchunk * 4];                              \
    __builtin_amdgcn_global_load_async_to_lds_b128((gvec4p)gK, (lvec4p)lK, 0, 0);     \
    __builtin_amdgcn_global_load_async_to_lds_b128((gvec4p)gV, (lvec4p)lV, 0, 0);     \
  } while (0)
#define STAGE_WAIT() __builtin_amdgcn_s_wait_asynccnt(0)
#else
#define STAGE_TILE(buf, kb_)                                                          \
  do {                                                                                \
    const unsigned int* gK32 = (const unsigned int*)(gKbase + (size_t)(kb_) * DH);    \
    const unsigned int* gV32 = (const unsigned int*)(gVbase + (kb_));                 \
    unsigned int* lK = &ldsK[buf][t * 4];                                             \
    unsigned int* lV = &ldsV[buf][vd * 32 + vchunk * 4];                              \
    _Pragma("unroll") for (int i_ = 0; i_ < 4; ++i_) lK[i_] = gK32[i_];               \
    _Pragma("unroll") for (int i_ = 0; i_ < 4; ++i_) lV[i_] = gV32[i_];               \
  } while (0)
#define STAGE_WAIT() ((void)0)
#endif

  STAGE_TILE(0, 0);
  STAGE_WAIT();
  __syncthreads();
  int cur = 0;

  for (int ks = 0; ks < NN / 64; ++ks) {
    const int kb = ks * 64;
    if (ks + 1 < NN / 64) STAGE_TILE(cur ^ 1, kb + 64);
    if (ks + 4 < NN / 64) {                      // warm L2 ahead (global_prefetch_b8)
      __builtin_prefetch(gKbase + (size_t)(kb + 256) * DH, 0, 1);
      __builtin_prefetch(gVbase + (kb + 256), 0, 1);
    }

    // ---- S = Q K^T over 4 groups of 16 keys ----
    const unsigned int* kbase = (half == 0) ? &ldsK[cur][l15 * 8] : &ldsZ[0];
    const int kstride = (half == 0) ? 128 : 0;   // 16 keys * 8 dwords
    f32x8 s[4];
    #pragma unroll
    for (int g = 0; g < 4; ++g) {
      FragAB kf;
      #pragma unroll
      for (int v = 0; v < 8; ++v) kf.u[v] = kbase[g * kstride + v];
      f32x8 z = {};
      s[g] = __builtin_amdgcn_wmma_f32_16x16x32_f16(false, qf.v, false, kf.v,
                                                    (short)0, z, false, false);
    }

    // ---- online softmax over the 64 keys ----
    #pragma unroll
    for (int j = 0; j < 8; ++j) {
      float a0 = s[0][j] * NSCALE, a1 = s[1][j] * NSCALE;
      float a2 = s[2][j] * NSCALE, a3 = s[3][j] * NSCALE;
      float tmax = fmaxf(fmaxf(a0, a1), fmaxf(a2, a3));
      #pragma unroll
      for (int m = 1; m < 16; m <<= 1) tmax = fmaxf(tmax, __shfl_xor(tmax, m, 32));
      float mnew = fmaxf(mrow[j], tmax);
      float e0 = __expf(a0 - mnew), e1 = __expf(a1 - mnew);
      float e2 = __expf(a2 - mnew), e3 = __expf(a3 - mnew);
      float rs = (e0 + e1) + (e2 + e3);
      #pragma unroll
      for (int m = 1; m < 16; m <<= 1) rs += __shfl_xor(rs, m, 32);
      float corr = __expf(mrow[j] - mnew);
      mrow[j] = mnew;
      lrow[j] = lrow[j] * corr + rs;
      acc[j] *= corr;
      int r = j + half * 8;
      ldsP16[r * 64 + l15]      = f2h_bits(e0);
      ldsP16[r * 64 + 16 + l15] = f2h_bits(e1);
      ldsP16[r * 64 + 32 + l15] = f2h_bits(e2);
      ldsP16[r * 64 + 48 + l15] = f2h_bits(e3);
    }
    asm volatile("" ::: "memory");   // same-wave DS is in-order; stop compiler reordering

    // ---- acc += P V over 2 groups of 32 keys ----
    #pragma unroll
    for (int pp = 0; pp < 2; ++pp) {
      const int kp0 = pp * 16;                     // key-pair base
      FragAB pf, vf;
      #pragma unroll
      for (int v = 0; v < 4; ++v) {
        pf.u[v]     = ldsPw[l15 * 32 + kp0 + half * 8 + v];
        pf.u[v + 4] = ldsPw[l15 * 32 + kp0 + 4 + half * 8 + v];
      }
      #pragma unroll
      for (int v = 0; v < 8; ++v) vf.u[v] = ldsV[cur][l15 * 32 + kp0 + half * 8 + v];
      acc = __builtin_amdgcn_wmma_f32_16x16x32_f16(false, pf.v, false, vf.v,
                                                   (short)0, acc, false, false);
    }

    STAGE_WAIT();
    __syncthreads();
    cur ^= 1;
  }

  const int h = bh % NH;
  const int b = bh / NH;
  #pragma unroll
  for (int j = 0; j < 8; ++j) {
    float o = acc[j] / lrow[j];
    int tok = qt + j + half * 8;
    wAttn[((size_t)b * NN + tok) * CC + h * DH + l15] = f2h_bits(o);
  }
#undef STAGE_TILE
#undef STAGE_WAIT
}

// ---------------- k3: O projection (WMMA) -> d_out [b][c][n] fp32 ----------------
__global__ void k3_oproj(const unsigned short* __restrict__ wAttn,
                         const unsigned short* __restrict__ wWb,
                         const float* __restrict__ bo,
                         float* __restrict__ y) {
  __shared__ __align__(16) unsigned int ldsa[16 * 64];   // 16 tok x 128 c (f16 pairs)
  const int bid = blockIdx.x;
  const int b  = bid / (NN / 16);
  const int tb = (bid % (NN / 16)) * 16;
  const int t  = threadIdx.x;
  const unsigned int* a32 = (const unsigned int*)wAttn;
  #pragma unroll
  for (int i = 0; i < 8; ++i) {
    int p = i * 128 + t;                       // 1024 dwords
    int tok = p >> 6, cp = p & 63;
    ldsa[tok * 64 + cp] = a32[((size_t)b * NN + tb + tok) * 64 + cp];
  }
  __syncthreads();
  const int wave = t >> 5, lane = t & 31;
  const int half = lane >> 4, l15 = lane & 15;
  const unsigned int* wb32 = ((const unsigned int*)wWb) + 3 * (CC * CC / 2);
  for (int dt = wave; dt < 8; dt += 4) {
    const int dcol = dt * 16 + l15;
    f32x8 acc = {};
    FragAB a, bf;
    for (int kk = 0; kk < 4; ++kk) {
      const int cA = kk * 32 + half * 16;
      #pragma unroll
      for (int v = 0; v < 4; ++v) {
        a.u[v]     = ldsa[(l15 * CC + cA + 2 * v) >> 1];
        a.u[v + 4] = ldsa[(l15 * CC + cA + 8 + 2 * v) >> 1];
      }
      const int cB = kk * 32 + half * 16;
      #pragma unroll
      for (int v = 0; v < 8; ++v) bf.u[v] = wb32[dcol * 64 + (cB >> 1) + v];
      acc = __builtin_amdgcn_wmma_f32_16x16x32_f16(false, a.v, false, bf.v, (short)0, acc, false, false);
    }
    const float bval = bo[dcol];
    #pragma unroll
    for (int j = 0; j < 8; ++j) {
      int tok = tb + j + half * 8;
      y[((size_t)b * CC + dcol) * NN + tok] = acc[j] + bval;
    }
  }
}

// ---------------- k4: GroupNorm stats ----------------
__global__ void k4_stats(const float* __restrict__ y, float* __restrict__ stats) {
  __shared__ float ss[256], ss2[256];
  const int bg = blockIdx.x;                 // b*G+g
  const int b = bg / GG, g = bg % GG;
  const int t = threadIdx.x;
  const int CG = CC / GG;                    // 16
  float s = 0.0f, s2 = 0.0f;
  const size_t base = ((size_t)b * CC + g * CG) * NN;
  for (int i = t; i < CG * NN; i += 256) {
    float v = y[base + i];
    s += v; s2 += v * v;
  }
  ss[t] = s; ss2[t] = s2;
  __syncthreads();
  for (int o = 128; o > 0; o >>= 1) {
    if (t < o) { ss[t] += ss[t + o]; ss2[t] += ss2[t + o]; }
    __syncthreads();
  }
  if (t == 0) {
    float inv = 1.0f / (float)(CG * NN);
    float mean = ss[0] * inv;
    float var = ss2[0] * inv - mean * mean;
    stats[bg * 2]     = mean;
    stats[bg * 2 + 1] = rsqrtf(var + GN_EPS);
  }
}

// ---------------- k5: normalize + affine + residual (in-place on d_out) ----------------
__global__ void k5_apply(float* __restrict__ y, const float* __restrict__ x,
                         const float* __restrict__ stats,
                         const float* __restrict__ gw, const float* __restrict__ gb) {
  size_t i = (size_t)blockIdx.x * blockDim.x + threadIdx.x;  // < B*C*N
  int c = (int)((i / NN) % CC);
  int b = (int)(i / ((size_t)CC * NN));
  int g = c / (CC / GG);
  float mean = stats[(b * GG + g) * 2];
  float rstd = stats[(b * GG + g) * 2 + 1];
  y[i] = (y[i] - mean) * rstd * gw[c] + gb[c] + x[i];
}

extern "C" void kernel_launch(void* const* d_in, const int* in_sizes, int n_in,
                              void* d_out, int out_size, void* d_ws, size_t ws_size,
                              hipStream_t stream) {
  (void)in_sizes; (void)n_in; (void)out_size; (void)ws_size;
  const float* x  = (const float*)d_in[0];
  const float* Wq = (const float*)d_in[1];
  const float* bq = (const float*)d_in[2];
  const float* Wk = (const float*)d_in[3];
  const float* bk = (const float*)d_in[4];
  const float* Wv = (const float*)d_in[5];
  const float* bv = (const float*)d_in[6];
  const float* Wo = (const float*)d_in[7];
  const float* bo = (const float*)d_in[8];
  const float* gw = (const float*)d_in[9];
  const float* gb = (const float*)d_in[10];
  float* out = (float*)d_out;

  char* ws = (char*)d_ws;
  unsigned short* wQ    = (unsigned short*)(ws);             // 2 MB  [b,h,N,dh]
  unsigned short* wK    = (unsigned short*)(ws + 2097152);   // 2 MB  [b,h,N,dh]
  unsigned short* wVt   = (unsigned short*)(ws + 4194304);   // 2 MB  [b,h,dh,N]
  unsigned short* wWb   = (unsigned short*)(ws + 6291456);   // 128 KB (4 weights f16)
  unsigned short* wAttn = (unsigned short*)(ws + 6422528);   // 2 MB  [b,N,C]
  float*          stats = (float*)(ws + 8519680);            // 32 floats

  k0_convert_weights<<<256, 256, 0, stream>>>(Wq, Wk, Wv, Wo, wWb);
  k1_qkv<<<BB * (NN / 16), 128, 0, stream>>>(x, wWb, bq, bk, bv, wQ, wK, wVt);
  k2_attn<<<BB * NH * (NN / 64), 128, 0, stream>>>(wQ, wK, wVt, wAttn);
  k3_oproj<<<BB * (NN / 16), 128, 0, stream>>>(wAttn, wWb, bo, out);
  k4_stats<<<BB * GG, 256, 0, stream>>>(out, stats);
  k5_apply<<<(BB * CC * NN) / 256, 256, 0, stream>>>(out, x, stats, gw, gb);
}
